// DeltaHebbianBlock_39281770889408
// MI455X (gfx1250) — matrix operand order repaired and checked by reading, not compiled, not run on hardware
//
#include <hip/hip_runtime.h>
#include <hip/hip_bf16.h>

// ---------------- problem constants ----------------
#define TD     8192     // T
#define DM     1024     // D_MODEL
#define HD     256      // HEAD_DIM
#define NH     4        // heads
#define CH     64       // chunk
#define NCHUNK 128      // T / CH
#define NSPLIT 4        // d-column split of the scan state

// ---------------- CDNA5 feature detection (compile-safe guards) -----------
#if defined(__has_builtin)
#if __has_builtin(__builtin_amdgcn_tensor_load_to_lds) && \
    __has_builtin(__builtin_amdgcn_s_wait_tensorcnt)
#define USE_TDM 1
#endif
#if __has_builtin(__builtin_amdgcn_global_load_async_to_lds_b128) && \
    __has_builtin(__builtin_amdgcn_s_wait_asynccnt)
#define USE_ASYNC 1
#endif
#endif

typedef __attribute__((ext_vector_type(16))) __bf16          v16bf;
typedef __attribute__((ext_vector_type(16))) unsigned short  v16u;
typedef __attribute__((ext_vector_type(8)))  float           v8f;

#ifdef USE_ASYNC
// async builtin takes: (v4i AS1* gaddr, v4i AS3* ldsaddr, int offset, int cpol)
typedef int v4i_vec __attribute__((vector_size(16)));
typedef __attribute__((address_space(1))) v4i_vec* as1_v4i;
typedef __attribute__((address_space(3))) v4i_vec* as3_v4i;
#endif

__device__ __forceinline__ float bf2f(unsigned short u) {
  unsigned int v = ((unsigned int)u) << 16;
  return __builtin_bit_cast(float, v);
}
__device__ __forceinline__ unsigned short f2bf(float f) {
  unsigned int u = __builtin_bit_cast(unsigned int, f);
  u += 0x7FFFu + ((u >> 16) & 1u);          // round-to-nearest-even
  return (unsigned short)(u >> 16);
}
__device__ __forceinline__ v8f vzero8() {
  v8f z;
  #pragma unroll
  for (int i = 0; i < 8; ++i) z[i] = 0.0f;
  return z;
}
__device__ __forceinline__ v8f wmma_bf16(v16bf a, v16bf b, v8f c) {
  return __builtin_amdgcn_wmma_f32_16x16x32_bf16(false, a, false, b, (short)0, c,
                                                 false, false);
}

// raw 32-bit LDS byte offset of a __shared__ object (via AS3 cast)
__device__ __forceinline__ unsigned lds_off(const void* p) {
  return (unsigned)(size_t)(const __attribute__((address_space(3))) void*)p;
}

// ---- fragment builders (contiguous k = 16*half + s, wave32) ----
__device__ __forceinline__ int frag_kbase(int i, int half) {
  return half * 16 + 2 * i;
}

// A(m0+(lane&15), k0+k) from row-major bf16 smem
__device__ __forceinline__ v16bf frag_a(const unsigned short* s, int ld, int m0,
                                        int k0, int lane) {
  const unsigned short* row = s + (size_t)(m0 + (lane & 15)) * ld + k0;
  int half = lane >> 4;
  v16u t;
  #pragma unroll
  for (int i = 0; i < 8; ++i) {
    int kb = frag_kbase(i, half);
    t[2 * i]     = row[kb];
    t[2 * i + 1] = row[kb + 1];
  }
  return __builtin_bit_cast(v16bf, t);
}

// A with per-lane row scale (rk_g = rk * gamma^row)
__device__ __forceinline__ v16bf frag_a_rowscale(const unsigned short* s, int ld,
                                                 int m0, int k0, int lane,
                                                 float scale) {
  const unsigned short* row = s + (size_t)(m0 + (lane & 15)) * ld + k0;
  int half = lane >> 4;
  v16u t;
  #pragma unroll
  for (int i = 0; i < 8; ++i) {
    int kb = frag_kbase(i, half);
    t[2 * i]     = f2bf(bf2f(row[kb]) * scale);
    t[2 * i + 1] = f2bf(bf2f(row[kb + 1]) * scale);
  }
  return __builtin_bit_cast(v16bf, t);
}

// A(m,k) = s[k][m] * gp[63-k]  (wk_g = diag-scaled transpose of wk chunk)
__device__ __forceinline__ v16bf frag_aT_kscale(const unsigned short* s, int ld,
                                                int m0, int k0, int lane,
                                                const float* gp) {
  int m = m0 + (lane & 15);
  int half = lane >> 4;
  v16u t;
  #pragma unroll
  for (int i = 0; i < 8; ++i) {
    int k = k0 + frag_kbase(i, half);
    t[2 * i]     = f2bf(bf2f(s[(size_t)k * ld + m]) * gp[63 - k]);
    t[2 * i + 1] = f2bf(bf2f(s[(size_t)(k + 1) * ld + m]) * gp[63 - (k + 1)]);
  }
  return __builtin_bit_cast(v16bf, t);
}

// B(k0+k, n0+(lane&15)) from row-major (K-major) source: strided
__device__ __forceinline__ v16bf frag_b(const unsigned short* s, int ld, int k0,
                                        int n0, int lane) {
  int n = n0 + (lane & 15);
  int kb = k0 + (lane >> 4) * 16;
  v16u t;
  #pragma unroll
  for (int i = 0; i < 8; ++i) {
    t[2 * i]     = s[(size_t)(kb + 2 * i) * ld + n];
    t[2 * i + 1] = s[(size_t)(kb + 2 * i + 1) * ld + n];
  }
  return __builtin_bit_cast(v16bf, t);
}

// B(k,n) = s[n][k] : transposed source -> contiguous 2x b128 per lane
__device__ __forceinline__ v16bf frag_bT(const unsigned short* s, int ld, int k0,
                                         int n0, int lane) {
  const unsigned short* row = s + (size_t)(n0 + (lane & 15)) * ld + k0 + (lane >> 4) * 16;
  v16u t;
  #pragma unroll
  for (int i = 0; i < 8; ++i) {
    t[2 * i]     = row[2 * i];
    t[2 * i + 1] = row[2 * i + 1];
  }
  return __builtin_bit_cast(v16bf, t);
}

// ---------------- TDM descriptor build + issue (ISA ch.8) ----------------
#ifdef USE_TDM
typedef __attribute__((ext_vector_type(4))) unsigned int u32x4;
typedef __attribute__((ext_vector_type(8))) int          i32x8;
typedef __attribute__((ext_vector_type(4))) int          i32x4;

// 2-D tile load: rows x tile0 elements of 2 bytes, tensor row stride stride0
__device__ __forceinline__ void tdm_load_2d(unsigned ldsByte, const void* gptr,
                                            unsigned dim0, unsigned rows,
                                            unsigned tile0, unsigned tile1,
                                            unsigned stride0) {
  unsigned long long ga = (unsigned long long)gptr;
  u32x4 g0;
  g0[0] = 1u;                                              // count=1 (valid user D#)
  g0[1] = ldsByte;                                         // lds_addr
  g0[2] = (unsigned)(ga & 0xffffffffu);                    // global_addr[31:0]
  g0[3] = (unsigned)((ga >> 32) & 0x01ffffffu) | 0x80000000u;  // [56:32] | type=2
  i32x8 g1;
  g1[0] = 0x10000;                                         // data_size=1 (2 bytes)
  g1[1] = (int)((dim0 & 0xffffu) << 16);                   // tensor_dim0[15:0]
  g1[2] = (int)((dim0 >> 16) | ((rows & 0xffffu) << 16));  // dim0[31:16]|dim1[15:0]
  g1[3] = (int)(((rows >> 16) & 0xffffu) | (tile0 << 16)); // dim1[31:16]|tile_dim0
  g1[4] = (int)(tile1 & 0xffffu);                          // tile_dim1, tile_dim2=0
  g1[5] = (int)stride0;                                    // tensor_dim0_stride
  g1[6] = 0;
  g1[7] = 0;
  i32x4 z4;
  z4[0] = z4[1] = z4[2] = z4[3] = 0;
#if __clang_major__ >= 23
  i32x8 z8;
  #pragma unroll
  for (int i = 0; i < 8; ++i) z8[i] = 0;
  __builtin_amdgcn_tensor_load_to_lds(g0, g1, z4, z4, z8, 0);
#else
  __builtin_amdgcn_tensor_load_to_lds(g0, g1, z4, z4, 0);
#endif
}
#endif

// ---------------- kernel 1: weights fp32 -> bf16 ----------------
__global__ void k_convert_w(const float* __restrict__ Ww, const float* __restrict__ Wr,
                            unsigned short* __restrict__ wwb,
                            unsigned short* __restrict__ wrb) {
  int i = blockIdx.x * 256 + threadIdx.x;   // grid covers DM*DM exactly
  wwb[i] = f2bf(Ww[i]);
  wrb[i] = f2bf(Wr[i]);
}

// ---------------- kernel 2: bf16 copy of x + per-head L2-normalized rk -----
__global__ void __launch_bounds__(256) k_prep(const float* __restrict__ x,
                                              unsigned short* __restrict__ xb,
                                              unsigned short* __restrict__ rkg) {
  int row = blockIdx.x;
  int tid = threadIdx.x;
  const float* xr = x + (size_t)row * DM;
  float4 v = ((const float4*)xr)[tid];

  __shared__ float red[256];
  __shared__ float nrm[NH];
  red[tid] = v.x * v.x + v.y * v.y + v.z * v.z + v.w * v.w;
  __syncthreads();
  if (tid < NH) {
    float s = 0.0f;
    #pragma unroll 4
    for (int i = 0; i < 64; ++i) s += red[tid * 64 + i];
    nrm[tid] = fmaxf(sqrtf(s), 1e-12f);
  }
  __syncthreads();
  int h = tid >> 6;
  float inv = 1.0f / nrm[h];

  unsigned short* xo = xb + (size_t)row * DM + tid * 4;
  xo[0] = f2bf(v.x); xo[1] = f2bf(v.y); xo[2] = f2bf(v.z); xo[3] = f2bf(v.w);

  int b = row >> 13, t = row & (TD - 1);
  int col = (tid & 63) * 4;
  unsigned short* ro = rkg + ((size_t)(b * NH + h) * TD + t) * HD + col;
  ro[0] = f2bf(v.x * inv); ro[1] = f2bf(v.y * inv);
  ro[2] = f2bf(v.z * inv); ro[3] = f2bf(v.w * inv);
}

// ---------------- kernel 3/5: C = A(16384x1024) * W^T, bf16 WMMA -----------
template <int MODE>
__global__ void __launch_bounds__(128) k_gemm(const unsigned short* __restrict__ A,
                                              const unsigned short* __restrict__ Bw,
                                              unsigned short* __restrict__ vout,
                                              const float* __restrict__ base,
                                              float* __restrict__ fout) {
  (void)vout; (void)base; (void)fout;
  __shared__ __align__(16) unsigned short As[64 * 32];
  __shared__ __align__(16) unsigned short Bs[64 * 32];

  int tid = threadIdx.x, lane = tid & 31, wv = tid >> 5;
  int mBase = blockIdx.x * 64;
  int nBase = blockIdx.y * 64;

  v8f acc[4];
  #pragma unroll
  for (int t = 0; t < 4; ++t) acc[t] = vzero8();

  for (int kk = 0; kk < DM; kk += 32) {
#ifdef USE_ASYNC
    for (int idx = tid; idx < 64 * 4; idx += 128) {
      int r = idx >> 2, c = idx & 3;
      __builtin_amdgcn_global_load_async_to_lds_b128(
          (as1_v4i)(void*)(A + (size_t)(mBase + r) * DM + kk + c * 8),
          (as3_v4i)(void*)&As[(size_t)idx * 8], 0, 0);
      __builtin_amdgcn_global_load_async_to_lds_b128(
          (as1_v4i)(void*)(Bw + (size_t)(nBase + r) * DM + kk + c * 8),
          (as3_v4i)(void*)&Bs[(size_t)idx * 8], 0, 0);
    }
    __builtin_amdgcn_s_wait_asynccnt(0);
#else
    for (int idx = tid; idx < 64 * 4; idx += 128) {
      int r = idx >> 2, c = idx & 3;
      ((uint4*)As)[idx] = ((const uint4*)(A + (size_t)(mBase + r) * DM + kk))[c];
    }
    for (int idx = tid; idx < 64 * 4; idx += 128) {
      int r = idx >> 2, c = idx & 3;
      ((uint4*)Bs)[idx] = ((const uint4*)(Bw + (size_t)(nBase + r) * DM + kk))[c];
    }
#endif
    __syncthreads();
    v16bf af = frag_a(As, 32, wv * 16, 0, lane);
    #pragma unroll
    for (int t = 0; t < 4; ++t) {
      v16bf bf = frag_bT(Bs, 32, 0, t * 16, lane);   // B(k,n)=W[n][k]
      acc[t] = wmma_bf16(af, bf, acc[t]);
    }
    __syncthreads();
  }

  #pragma unroll
  for (int t = 0; t < 4; ++t) {
    #pragma unroll
    for (int r = 0; r < 8; ++r) {
      int m = mBase + wv * 16 + r + ((lane >> 4) << 3);
      int n = nBase + t * 16 + (lane & 15);
      float val = acc[t][r];
      if (MODE == 0) {
        int b = m >> 13, tt = m & (TD - 1), h = n >> 8, j = n & (HD - 1);
        vout[((size_t)(b * NH + h) * TD + tt) * HD + j] = f2bf(val);
      } else {
        size_t idx = (size_t)m * DM + n;
        fout[idx] = base[idx] + val;
      }
    }
  }
}

// ---------------- kernel 4: chunked scan (one (bh, d-slice) per block) -----
__global__ void __launch_bounds__(256) k_scan(const unsigned short* __restrict__ vg,
                                              const unsigned short* __restrict__ rkg,
                                              const float* __restrict__ decay,
                                              const float* __restrict__ log_alpha,
                                              unsigned short* __restrict__ ob) {
  __shared__ __align__(16) unsigned short sSt[64 * HD];   // S^T : [n(64)][k(256)] bf16
  __shared__ __align__(16) unsigned short sRKW[65 * HD];  // row c: wk[c]; rk[c]=row c+1
  __shared__ __align__(16) unsigned short sV[64 * 64];    // v chunk slice (row-major)
  __shared__ __align__(16) unsigned short sVNt[64 * 64];  // v_new^T : [n][k]
  __shared__ __align__(16) unsigned short sWC[64 * HD];   // wk_corr; aliased as f32 T
  __shared__ __align__(16) unsigned short sXb[64 * 64];   // Ainv bf16
  __shared__ __align__(16) unsigned short sTb[64 * 64];   // intra bf16
  __shared__ float sX[64 * 64];                           // Ainv fp32 (substitution)
  __shared__ float sGP[65];                               // gamma^p

  float* sT = (float*)sWC;  // T = I - A0 (fp32), lifetime disjoint from wk_corr

  int tid = threadIdx.x, lane = tid & 31, wv = tid >> 5;
  int bh = blockIdx.x >> 2;            // / NSPLIT
  int sp = blockIdx.x & (NSPLIT - 1);
  int col0 = sp * 64;
  int b = bh >> 2, h = bh & (NH - 1);

  float gamma = 1.0f / (1.0f + __expf(-decay[h]));
  float lg = __logf(gamma);
  float alpha = __expf(log_alpha[h]);
  if (tid < 65) sGP[tid] = __expf(lg * (float)tid);
  for (int i = tid; i < 64 * HD / 8; i += 256) ((uint4*)sSt)[i] = make_uint4(0, 0, 0, 0);
  __syncthreads();
  float gC = sGP[64];

  const size_t rowBase = (size_t)bh * TD;

  for (int cn = 0; cn < NCHUNK; ++cn) {
    int t0 = cn * CH;

    // ---- chunk loads: TDM 2-D tiles (v slice + shifted rk/wk panel) ----
#ifdef USE_TDM
    if (wv == 0) {
      tdm_load_2d(lds_off(sV), vg + (rowBase + t0) * HD + col0,
                  /*dim0*/64, /*rows*/64, /*tile0*/64, /*tile1*/64, /*stride0*/HD);
      if (cn == 0)
        tdm_load_2d(lds_off(sRKW) + HD * 2, rkg + rowBase * HD,
                    HD, 64, HD, 64, HD);
      else
        tdm_load_2d(lds_off(sRKW), rkg + (rowBase + t0 - 1) * HD,
                    HD, 65, HD, 65, HD);
    }
    if (cn == 0 && tid < 32) ((uint4*)sRKW)[tid] = make_uint4(0, 0, 0, 0);
    if (cn + 1 < NCHUNK && tid == 0) {
      __builtin_prefetch(vg + (rowBase + t0 + CH) * HD + col0, 0, 1);
      __builtin_prefetch(rkg + (rowBase + t0 + CH) * HD, 0, 1);
    }
    if (wv == 0) __builtin_amdgcn_s_wait_tensorcnt(0);
#else
    for (int idx = tid; idx < 64 * 8; idx += 256) {
      int r = idx >> 3, c = idx & 7;
      ((uint4*)sV)[r * 8 + c] =
          ((const uint4*)(vg + (rowBase + t0 + r) * HD + col0))[c];
    }
    for (int idx = tid; idx < 64 * 32; idx += 256) {
      int r = idx >> 5, c = idx & 31;
      ((uint4*)sRKW)[(r + 1) * 32 + c] =
          ((const uint4*)(rkg + (rowBase + t0 + r) * HD))[c];
    }
    if (tid < 32) {
      uint4 z = make_uint4(0, 0, 0, 0);
      ((uint4*)sRKW)[tid] =
          (cn == 0) ? z : ((const uint4*)(rkg + (rowBase + t0 - 1) * HD))[tid];
    }
    if (cn + 1 < NCHUNK && tid == 0) {
      __builtin_prefetch(vg + (rowBase + t0 + CH) * HD + col0, 0, 1);
      __builtin_prefetch(rkg + (rowBase + t0 + CH) * HD, 0, 1);
    }
#endif
    __syncthreads();

    // ---- Phase A: T = I + strict(wk wk^T * L), 16 tiles / 8 waves ----
    #pragma unroll
    for (int q = 0; q < 2; ++q) {
      int tI = wv * 2 + q, ti = tI >> 2, tj = tI & 3;
      v8f acc = vzero8();
      for (int k = 0; k < HD; k += 32)
        acc = wmma_bf16(frag_a(sRKW, HD, ti * 16, k, lane),
                        frag_bT(sRKW, HD, k, tj * 16, lane), acc);
      #pragma unroll
      for (int r = 0; r < 8; ++r) {
        int i = ti * 16 + r + ((lane >> 4) << 3);
        int j = tj * 16 + (lane & 15);
        sT[i * 64 + j] = (i == j) ? 1.0f : ((i > j) ? acc[r] * sGP[i - j] : 0.0f);
      }
    }
    __syncthreads();

    // ---- Phase B: forward substitution, X = inv(T) (unit lower) ----
    {
      int j = tid;
      for (int i = 0; i < 64; ++i) {
        if (j < 64) {
          float s = (i == j) ? 1.0f : 0.0f;
          for (int k = j; k < i; ++k) s -= sT[i * 64 + k] * sX[k * 64 + j];
          sX[i * 64 + j] = s;
        }
        __syncthreads();
      }
    }

    // ---- Phase E: intra = strict(rk wk^T * L) -> bf16; also X -> bf16 ----
    #pragma unroll
    for (int q = 0; q < 2; ++q) {
      int tI = wv * 2 + q, ti = tI >> 2, tj = tI & 3;
      v8f acc = vzero8();
      for (int k = 0; k < HD; k += 32)
        acc = wmma_bf16(frag_a(sRKW + HD, HD, ti * 16, k, lane),  // rk rows
                        frag_bT(sRKW, HD, k, tj * 16, lane), acc);
      #pragma unroll
      for (int r = 0; r < 8; ++r) {
        int i = ti * 16 + r + ((lane >> 4) << 3);
        int j = tj * 16 + (lane & 15);
        sTb[i * 64 + j] = f2bf((i > j) ? acc[r] * sGP[i - j] : 0.0f);
      }
    }
    for (int i = tid; i < 64 * 64; i += 256) sXb[i] = f2bf(sX[i]);
    __syncthreads();

    // ---- Phase C: wk_corr = X @ wk (64 x 256), 64 tiles / 8 waves ----
    #pragma unroll
    for (int q = 0; q < 8; ++q) {
      int tI = wv * 8 + q, ti = tI >> 4, tj = tI & 15;
      v8f acc = vzero8();
      for (int k = 0; k < 64; k += 32)
        acc = wmma_bf16(frag_a(sXb, 64, ti * 16, k, lane),
                        frag_b(sRKW, HD, k, tj * 16, lane), acc);
      #pragma unroll
      for (int r = 0; r < 8; ++r) {
        int i = ti * 16 + r + ((lane >> 4) << 3);
        int j = tj * 16 + (lane & 15);
        sWC[i * HD + j] = f2bf(acc[r]);
      }
    }
    __syncthreads();

    // ---- Phase D: v_new = X@v - wk_corr@S (64x64 slice), write transposed --
    #pragma unroll
    for (int q = 0; q < 2; ++q) {
      int tI = wv * 2 + q, ti = tI >> 2, tj = tI & 3;
      v8f acc = vzero8();
      for (int k = 0; k < 64; k += 32)
        acc = wmma_bf16(frag_a(sXb, 64, ti * 16, k, lane),
                        frag_b(sV, 64, k, tj * 16, lane), acc);
      v8f acc2 = vzero8();
      for (int k = 0; k < HD; k += 32)
        acc2 = wmma_bf16(frag_a(sWC, HD, ti * 16, k, lane),
                         frag_bT(sSt, HD, k, tj * 16, lane), acc2);
      #pragma unroll
      for (int r = 0; r < 8; ++r) {
        int i = ti * 16 + r + ((lane >> 4) << 3);
        int j = tj * 16 + (lane & 15);
        sVNt[j * 64 + i] = f2bf(acc[r] - acc2[r]);   // contiguous in r per lane
      }
    }
    __syncthreads();

    // ---- Phase F: o = alpha*(rk_g@S + intra@v_new) -> global bf16 ----
    #pragma unroll
    for (int q = 0; q < 2; ++q) {
      int tI = wv * 2 + q, ti = tI >> 2, tj = tI & 3;
      float rowscale = sGP[ti * 16 + (lane & 15)];
      v8f acc = vzero8();
      for (int k = 0; k < HD; k += 32)
        acc = wmma_bf16(frag_a_rowscale(sRKW + HD, HD, ti * 16, k, lane, rowscale),
                        frag_bT(sSt, HD, k, tj * 16, lane), acc);
      for (int k = 0; k < 64; k += 32)
        acc = wmma_bf16(frag_a(sTb, 64, ti * 16, k, lane),
                        frag_bT(sVNt, 64, k, tj * 16, lane), acc);
      #pragma unroll
      for (int r = 0; r < 8; ++r) {
        int i = ti * 16 + r + ((lane >> 4) << 3);
        int j = tj * 16 + (lane & 15);
        int t = t0 + i;
        ob[(size_t)(b * TD + t) * DM + h * HD + col0 + j] = f2bf(alpha * acc[r]);
      }
    }
    __syncthreads();   // all reads of S done before update

    // ---- Phase G: S^T = gamma^C * S^T + (wk_g @ v_new)^T (256x64 slice) ----
    #pragma unroll
    for (int q = 0; q < 8; ++q) {
      int tI = wv * 8 + q, ti = tI >> 2, tj = tI & 3;   // ti over d/16, tj over 64/16
      v8f acc = vzero8();
      for (int k = 0; k < 64; k += 32)
        acc = wmma_bf16(frag_aT_kscale(sRKW, HD, ti * 16, k, lane, sGP),
                        frag_bT(sVNt, 64, k, tj * 16, lane), acc);
      #pragma unroll
      for (int r = 0; r < 8; ++r) {
        int i = ti * 16 + r + ((lane >> 4) << 3);       // d index (K of S^T)
        int j = tj * 16 + (lane & 15);                  // slice col (row of S^T)
        float old = bf2f(sSt[j * HD + i]);
        sSt[j * HD + i] = f2bf(gC * old + acc[r]);      // contiguous in r per lane
      }
    }
    __syncthreads();
  }
}

// ---------------- host-side launch ----------------
extern "C" void kernel_launch(void* const* d_in, const int* in_sizes, int n_in,
                              void* d_out, int out_size, void* d_ws, size_t ws_size,
                              hipStream_t stream) {
  (void)in_sizes; (void)n_in; (void)out_size; (void)ws_size;
  const float* x   = (const float*)d_in[0];
  const float* Ww  = (const float*)d_in[1];
  const float* Wr  = (const float*)d_in[2];
  const float* dec = (const float*)d_in[3];
  const float* la  = (const float*)d_in[4];
  float* outp = (float*)d_out;

  unsigned short* xb  = (unsigned short*)d_ws;              // 16M  x bf16
  unsigned short* vg  = xb  + (size_t)16384 * 1024;         // 16M  v (BH,T,HD)
  unsigned short* rkg = vg  + (size_t)8 * 8192 * 256;       // 16M  rk (BH,T,HD)
  unsigned short* ob  = rkg + (size_t)8 * 8192 * 256;       // 16M  o (B,T,D)
  unsigned short* wwb = ob  + (size_t)16384 * 1024;         // 1M   Ww bf16
  unsigned short* wrb = wwb + (size_t)1024 * 1024;          // 1M   Wr bf16

  k_convert_w<<<dim3(4096), dim3(256), 0, stream>>>(Ww, Wr, wwb, wrb);
  k_prep<<<dim3(16384), dim3(256), 0, stream>>>(x, xb, rkg);
  dim3 gg(256, 16);
  k_gemm<0><<<gg, dim3(128), 0, stream>>>(xb, wwb, vg, nullptr, nullptr);
  k_scan<<<dim3(32), dim3(256), 0, stream>>>(vg, rkg, dec, la, ob);
  k_gemm<1><<<gg, dim3(128), 0, stream>>>(ob, wrb, nullptr, x, outp);
}
